// SetGNN_11235634446340
// MI455X (gfx1250) — compile-verified, matrix operand order checked by reading, not run on hardware
//
#include <hip/hip_runtime.h>
#include <cstdint>
#include <math.h>

// ---------------------------------------------------------------------------
// SetGNN (AllSet PMA x4) forward for MI455X (gfx1250, wave32, WMMA).
// Dense GEMMs: bf16 WMMA (v_wmma_f32_16x16x32_bf16), fp32 accumulate,
// double-buffered LDS staging via GLOBAL_LOAD_ASYNC_TO_LDS_B128 (ASYNCcnt).
// alpha = K . att_r folded into a GEMV against wkr = WK @ att_r.
// Segment softmax uses L2-resident float atomics (agg <= 51MB << 192MB L2).
// ---------------------------------------------------------------------------

typedef __attribute__((ext_vector_type(16))) __bf16 bf16x16;
typedef __attribute__((ext_vector_type(8)))  float  fx8;

__device__ __forceinline__ uint32_t f2bf1(float f) {
  uint32_t x = __float_as_uint(f);
  uint32_t r = x + 0x7FFFu + ((x >> 16) & 1u);   // round-to-nearest-even
  return r >> 16;
}
__device__ __forceinline__ uint32_t packbf(float a, float b) {
  return f2bf1(a) | (f2bf1(b) << 16);
}
__device__ __forceinline__ float wave_sum(float v) {
#pragma unroll
  for (int m = 16; m >= 1; m >>= 1) v += __shfl_xor(v, m, 32);
  return v;
}
// ordered-int trick: works for any sign once buffer is initialized to -inf
__device__ __forceinline__ void atomicMaxF(float* addr, float val) {
  if (val >= 0.0f) atomicMax((int*)addr, __float_as_int(val));
  else             atomicMin((unsigned int*)addr, __float_as_uint(val));
}
// CDNA5 async copy: memory -> LDS, tracked by ASYNCcnt (no VGPR round trip)
__device__ __forceinline__ void async_copy_b128(uint32_t lds_off, const uint32_t* gsrc) {
  asm volatile("global_load_async_to_lds_b128 %0, %1, off"
               :: "v"(lds_off), "v"((uint64_t)(size_t)gsrc) : "memory");
}
__device__ __forceinline__ void async_wait0() {
  asm volatile("s_wait_asynccnt 0" ::: "memory");
}

// ------------------------------- small utils -------------------------------

__global__ void fill_f32(float* __restrict__ p, float v, int n) {
  int i = blockIdx.x * 256 + threadIdx.x;
  if (i < n) p[i] = v;
}
__global__ void fill_i32(int* __restrict__ p, int v, int n) {
  int i = blockIdx.x * 256 + threadIdx.x;
  if (i < n) p[i] = v;
}
__global__ void min_i32(const int* __restrict__ d, int* __restrict__ out, int n) {
  int i = blockIdx.x * 256 + threadIdx.x;
  if (i < n) atomicMin(out, d[i]);
}
__global__ void sub_i32(const int* __restrict__ d, const int* __restrict__ mn,
                        int* __restrict__ out, int n) {
  int i = blockIdx.x * 256 + threadIdx.x;
  if (i < n) out[i] = d[i] - mn[0];
}
// fp32 row-major -> packed bf16 pairs (dwords). n4 = #float4 groups.
__global__ void pack_acts(const float* __restrict__ X, uint32_t* __restrict__ Xb, int n4) {
  int i = blockIdx.x * 256 + threadIdx.x;
  if (i >= n4) return;
  float4 v = ((const float4*)X)[i];
  uint2 o;
  o.x = packbf(v.x, v.y);
  o.y = packbf(v.z, v.w);
  ((uint2*)Xb)[i] = o;
}
// weight (Kd,128) fp32 -> fragment-major packed bf16 dwords Wp[n][k2]
// (n = output column 0..127, k2 = K-pair index 0..Kd2-1)
__global__ void pack_w(const float* __restrict__ W, uint32_t* __restrict__ Wp, int Kd2) {
  int i = blockIdx.x * 256 + threadIdx.x;
  if (i >= Kd2 * 128) return;
  int n = i / Kd2, k2 = i % Kd2;
  float a = W[(size_t)(2 * k2) * 128 + n];
  float b = W[(size_t)(2 * k2 + 1) * 128 + n];
  Wp[i] = packbf(a, b);
}
// wkr[k] = dot(WK[k,:128], att_r); wkr[Kd] = dot(bK, att_r)
__global__ void make_wkr(const float* __restrict__ WK, const float* __restrict__ bK,
                         const float* __restrict__ att, float* __restrict__ wkr, int Kd) {
  int row = blockIdx.x * 8 + (threadIdx.x >> 5);
  int lane = threadIdx.x & 31;
  if (row > Kd) return;
  const float* v = (row == Kd) ? bK : (WK + (size_t)row * 128);
  float4 t = ((const float4*)v)[lane];
  float4 r = ((const float4*)att)[lane];
  float s = t.x * r.x + t.y * r.y + t.z * r.z + t.w * r.w;
  s = wave_sum(s);
  if (lane == 0) wkr[row] = s;
}
// a[m] = leaky_relu( H[m,:] . wkr + wkr[Kd] )
__global__ void gemv_alpha(const float* __restrict__ H, const float* __restrict__ wkr,
                           float* __restrict__ a, int M, int Kd) {
  int row = blockIdx.x * 8 + (threadIdx.x >> 5);
  int lane = threadIdx.x & 31;
  if (row >= M) return;
  const float4* hp = (const float4*)(H + (size_t)row * Kd);
  const float4* wp = (const float4*)wkr;
  float s = 0.0f;
  int nq = Kd >> 7;  // groups of 128 elements
  for (int j = 0; j < nq; ++j) {
    float4 h4 = hp[j * 32 + lane];
    float4 w4 = wp[j * 32 + lane];
    s += h4.x * w4.x + h4.y * w4.y + h4.z * w4.z + h4.w * w4.w;
  }
  s = wave_sum(s) + wkr[Kd];
  float av = (s > 0.0f) ? s : 0.2f * s;  // NEG_SLOPE = 0.2
  if (lane == 0) a[row] = av;
}

// ------------------------------ WMMA GEMM ----------------------------------
// C[M,128] = A(bf16 pairs, row-major [M][Kd2]) @ B(frag-major [128][Kd2]) + bias
// block = 256 thr (8 waves); BM=128; each wave owns a 16-row strip x 8 tiles.
// LDS tiles are fragment-major, stride 20 dwords so every fragment half is a
// 16B-aligned b128; staging is double-buffered async (ASYNCcnt).
__global__ __launch_bounds__(256)
void gemm_bf16_n128(const uint32_t* __restrict__ A, const uint32_t* __restrict__ B,
                    const float* __restrict__ bias, float* __restrict__ C,
                    int M, int Kd2, int relu) {
  __shared__ __align__(16) uint32_t As[2][128][20];  // [buf][m][k2]
  __shared__ __align__(16) uint32_t Bs[2][128][20];  // [buf][n][k2]
  int tid  = threadIdx.x;
  int bm   = blockIdx.x * 128;
  int w    = tid >> 5;
  int lane = tid & 31;
  int ln15 = lane & 15, hi = lane >> 4;

  fx8 acc[8];
#pragma unroll
  for (int t = 0; t < 8; ++t)
#pragma unroll
    for (int r = 0; r < 8; ++r) acc[t][r] = 0.0f;

  const int nk = Kd2 >> 4;  // K-steps of 32 elements (16 dwords)

  // staging assignment: thread covers row r (A: M-row, B: N-col), half hh
  const int r  = tid >> 1;
  const int hh = (tid & 1) * 8;
  const int gr = bm + r;
  const uint32_t* Asrc0 = A + (size_t)gr * Kd2 + hh;
  const uint32_t* Bsrc0 = B + (size_t)r * Kd2 + hh;

  auto stage = [&](int kk, int b) {
    if (gr < M) {
      const uint32_t* s = Asrc0 + kk * 16;
      uint32_t l = (uint32_t)(size_t)&As[b][r][hh];
      async_copy_b128(l, s);
      async_copy_b128(l + 16, s + 4);
      if (kk + 1 < nk) __builtin_prefetch(s + 16, 0, 1);  // global_prefetch_b8
    }
    {
      const uint32_t* s = Bsrc0 + kk * 16;
      uint32_t l = (uint32_t)(size_t)&Bs[b][r][hh];
      async_copy_b128(l, s);
      async_copy_b128(l + 16, s + 4);
    }
  };

  stage(0, 0);
  async_wait0();
  __syncthreads();

  int buf = 0;
  const int mrow = w * 16 + ln15;
  for (int kk = 0; kk < nk; ++kk) {
    if (kk + 1 < nk) stage(kk + 1, buf ^ 1);

    // A fragment: k2 = v + (v&4) + hi*4 -> two contiguous b128 halves
    union { uint4 q[2]; bf16x16 v; } af;
    af.q[0] = *(const uint4*)&As[buf][mrow][hi * 4];
    af.q[1] = *(const uint4*)&As[buf][mrow][8 + hi * 4];

#pragma unroll
    for (int t = 0; t < 8; ++t) {
      int n = t * 16 + ln15;
      union { uint4 q[2]; bf16x16 v; } bq;
      bq.q[0] = *(const uint4*)&Bs[buf][n][hi * 4];
      bq.q[1] = *(const uint4*)&Bs[buf][n][8 + hi * 4];
      acc[t] = __builtin_amdgcn_wmma_f32_16x16x32_bf16(
          false, af.v, false, bq.v, (short)0, acc[t], false, false);
    }

    async_wait0();     // next tile's async copies are complete
    __syncthreads();   // visible to all waves; LDS reads above already retired
    buf ^= 1;
  }

  // C/D layout: VGPR r, lanes 0-15 -> M=r, lanes 16-31 -> M=r+8; N = 16t+ln15
  int mbase = bm + w * 16 + hi * 8;
#pragma unroll
  for (int t = 0; t < 8; ++t) {
    int n = t * 16 + ln15;
    float bsv = bias[n];
#pragma unroll
    for (int rr = 0; rr < 8; ++rr) {
      int m = mbase + rr;
      if (m < M) {
        float v = acc[t][rr] + bsv;
        if (relu) v = fmaxf(v, 0.0f);
        C[(size_t)m * 128 + n] = v;
      }
    }
  }
}

// --------------------------- segment softmax -------------------------------

__global__ void seg_max(const int* __restrict__ s_idx, const int* __restrict__ d_idx,
                        const float* __restrict__ a, float* __restrict__ mx, int nnz) {
  int e = blockIdx.x * 256 + threadIdx.x;
  if (e >= nnz) return;
  atomicMaxF(&mx[d_idx[e]], a[s_idx[e]]);
}
__global__ void seg_den(const int* __restrict__ s_idx, const int* __restrict__ d_idx,
                        const float* __restrict__ a, const float* __restrict__ mx,
                        float* __restrict__ den, int nnz) {
  int e = blockIdx.x * 256 + threadIdx.x;
  if (e >= nnz) return;
  int d = d_idx[e];
  atomicAdd(&den[d], __expf(a[s_idx[e]] - mx[d]));
}
// one wave per incidence; lane covers 4 of 128 feature cols (float4 gather)
__global__ __launch_bounds__(256)
void scatter_v(const int* __restrict__ s_idx, const int* __restrict__ d_idx,
               const float* __restrict__ a, const float* __restrict__ mx,
               const float* __restrict__ den, const float* __restrict__ V,
               float* __restrict__ agg, int nnz) {
  int e = blockIdx.x * 8 + (threadIdx.x >> 5);
  int lane = threadIdx.x & 31;
  if (e >= nnz) return;
  int s = s_idx[e], d = d_idx[e];
  float ex = __expf(a[s] - mx[d]);
  float wgt = ex / fmaxf(den[d], 1e-16f);
  float4 v4 = ((const float4*)(V + (size_t)s * 128))[lane];
  float* ap = agg + (size_t)d * 128 + lane * 4;
  atomicAdd(ap + 0, v4.x * wgt);
  atomicAdd(ap + 1, v4.y * wgt);
  atomicAdd(ap + 2, v4.z * wgt);
  atomicAdd(ap + 3, v4.w * wgt);
}

// ------------------------------ epilogues ----------------------------------
// epiA: out = LN0(agg + att_r), in-place over agg, plus bf16-packed copy
__global__ void epiA(float* __restrict__ agg, const float* __restrict__ att,
                     const float* __restrict__ g, const float* __restrict__ b,
                     uint32_t* __restrict__ outb, int n_tgt) {
  int row = blockIdx.x * 8 + (threadIdx.x >> 5);
  int lane = threadIdx.x & 31;
  if (row >= n_tgt) return;
  float4 v = ((float4*)(agg + (size_t)row * 128))[lane];
  float4 ar = ((const float4*)att)[lane];
  v.x += ar.x; v.y += ar.y; v.z += ar.z; v.w += ar.w;
  float mean = wave_sum(v.x + v.y + v.z + v.w) * (1.0f / 128.0f);
  float dx = v.x - mean, dy = v.y - mean, dz = v.z - mean, dw = v.w - mean;
  float var = wave_sum(dx * dx + dy * dy + dz * dz + dw * dw) * (1.0f / 128.0f);
  float inv = rsqrtf(var + 1e-5f);
  float4 gg = ((const float4*)g)[lane];
  float4 bb = ((const float4*)b)[lane];
  float4 o;
  o.x = dx * inv * gg.x + bb.x;
  o.y = dy * inv * gg.y + bb.y;
  o.z = dz * inv * gg.z + bb.z;
  o.w = dw * inv * gg.w + bb.w;
  ((float4*)(agg + (size_t)row * 128))[lane] = o;
  uint2 p; p.x = packbf(o.x, o.y); p.y = packbf(o.z, o.w);
  ((uint2*)(outb + (size_t)row * 64))[lane] = p;
}
// epiB: h = relu(LN1(out + ff)); optional bf16-packed copy for next hop
__global__ void epiB(const float* __restrict__ out, const float* __restrict__ ff,
                     const float* __restrict__ g, const float* __restrict__ b,
                     float* __restrict__ h, uint32_t* __restrict__ hb, int n_tgt) {
  int row = blockIdx.x * 8 + (threadIdx.x >> 5);
  int lane = threadIdx.x & 31;
  if (row >= n_tgt) return;
  float4 v = ((const float4*)(out + (size_t)row * 128))[lane];
  float4 f = ((const float4*)(ff + (size_t)row * 128))[lane];
  v.x += f.x; v.y += f.y; v.z += f.z; v.w += f.w;
  float mean = wave_sum(v.x + v.y + v.z + v.w) * (1.0f / 128.0f);
  float dx = v.x - mean, dy = v.y - mean, dz = v.z - mean, dw = v.w - mean;
  float var = wave_sum(dx * dx + dy * dy + dz * dz + dw * dw) * (1.0f / 128.0f);
  float inv = rsqrtf(var + 1e-5f);
  float4 gg = ((const float4*)g)[lane];
  float4 bb = ((const float4*)b)[lane];
  float4 o;
  o.x = fmaxf(dx * inv * gg.x + bb.x, 0.0f);
  o.y = fmaxf(dy * inv * gg.y + bb.y, 0.0f);
  o.z = fmaxf(dz * inv * gg.z + bb.z, 0.0f);
  o.w = fmaxf(dw * inv * gg.w + bb.w, 0.0f);
  ((float4*)(h + (size_t)row * 128))[lane] = o;
  if (hb) {
    uint2 p; p.x = packbf(o.x, o.y); p.y = packbf(o.z, o.w);
    ((uint2*)(hb + (size_t)row * 64))[lane] = p;
  }
}

// ------------------------------- host side ---------------------------------

extern "C" void kernel_launch(void* const* d_in, const int* in_sizes, int n_in,
                              void* d_out, int out_size, void* d_ws, size_t ws_size,
                              hipStream_t stream) {
  (void)out_size; (void)ws_size;
  const int N_NODES = 100000, N_HEDGES = 25000;
  const int NNZ = in_sizes[n_in - 1] / 2;

  // input order: x, then params (v2e0,e2v0,v2e1,e2v1) x 13 leaves in dict
  // insertion order (WK,bK,WV,bV,att_r,ln0_g,ln0_b,f1W,f1b,f2W,f2b,ln1_g,ln1_b),
  // then edge_index.
  const float* x = (const float*)d_in[0];
  const int* ei = (const int*)d_in[n_in - 1];
  const int* e_src = ei;
  const int* e_dst = ei + NNZ;

  struct HP { const float *WK,*bK,*WV,*bV,*att,*ln0g,*ln0b,*f1W,*f1b,*f2W,*f2b,*ln1g,*ln1b; };
  HP P[4];
  for (int h = 0; h < 4; ++h) {
    const int base = 1 + 13 * h;
    P[h].WK   = (const float*)d_in[base + 0];
    P[h].bK   = (const float*)d_in[base + 1];
    P[h].WV   = (const float*)d_in[base + 2];
    P[h].bV   = (const float*)d_in[base + 3];
    P[h].att  = (const float*)d_in[base + 4];
    P[h].ln0g = (const float*)d_in[base + 5];
    P[h].ln0b = (const float*)d_in[base + 6];
    P[h].f1W  = (const float*)d_in[base + 7];
    P[h].f1b  = (const float*)d_in[base + 8];
    P[h].f2W  = (const float*)d_in[base + 9];
    P[h].f2b  = (const float*)d_in[base + 10];
    P[h].ln1g = (const float*)d_in[base + 11];
    P[h].ln1b = (const float*)d_in[base + 12];
  }

  // workspace carve (~405 MB)
  char* wp = (char*)d_ws;
  auto carve = [&](size_t bytes) -> void* {
    void* p = (void*)wp;
    wp += (bytes + 255) & ~(size_t)255;
    return p;
  };
  uint32_t* xb   = (uint32_t*)carve((size_t)N_NODES * 256 * 4);   // x bf16 pairs
  float*    hA   = (float*)carve((size_t)N_NODES * 128 * 4);
  uint32_t* hAb  = (uint32_t*)carve((size_t)N_NODES * 64 * 4);
  float*    hE   = (float*)carve((size_t)N_HEDGES * 128 * 4);
  uint32_t* hEb  = (uint32_t*)carve((size_t)N_HEDGES * 64 * 4);
  float*    Vff  = (float*)carve((size_t)N_NODES * 128 * 4);      // V, later ff
  float*    t1f  = (float*)carve((size_t)N_NODES * 128 * 4);
  uint32_t* t1b  = (uint32_t*)carve((size_t)N_NODES * 64 * 4);
  float*    agg  = (float*)carve((size_t)N_NODES * 128 * 4);      // agg -> out (in place)
  uint32_t* outb = (uint32_t*)carve((size_t)N_NODES * 64 * 4);
  float*    abuf = (float*)carve((size_t)N_NODES * 4);
  float*    mx   = (float*)carve((size_t)N_NODES * 4);
  float*    den  = (float*)carve((size_t)N_NODES * 4);
  int*      dsts = (int*)carve((size_t)NNZ * 4);
  int*      dmin = (int*)carve(256);
  float*    wkr  = (float*)carve((512 + 1) * 4);
  uint32_t* WVp  = (uint32_t*)carve((size_t)256 * 128 * 4);
  uint32_t* f1p  = (uint32_t*)carve((size_t)64 * 128 * 4);
  uint32_t* f2p  = (uint32_t*)carve((size_t)64 * 128 * 4);

  // prologue: pack x to bf16; dst shift = dst - min(dst)
  pack_acts<<<(N_NODES * 128 + 255) / 256, 256, 0, stream>>>(x, xb, N_NODES * 128);
  fill_i32<<<1, 1, 0, stream>>>(dmin, 0x7fffffff, 1);
  min_i32<<<(NNZ + 255) / 256, 256, 0, stream>>>(e_dst, dmin, NNZ);
  sub_i32<<<(NNZ + 255) / 256, 256, 0, stream>>>(e_dst, dmin, dsts, NNZ);

  auto hop = [&](const HP& p, const int* s_idx, const int* d_idx,
                 int n_src, int n_tgt, const float* Hf, const uint32_t* Hb,
                 int Kd, float* out_f, uint32_t* out_b) {
    const int Kd2 = Kd >> 1;
    pack_w<<<(Kd2 * 128 + 255) / 256, 256, 0, stream>>>(p.WV, WVp, Kd2);
    pack_w<<<(64 * 128 + 255) / 256, 256, 0, stream>>>(p.f1W, f1p, 64);
    pack_w<<<(64 * 128 + 255) / 256, 256, 0, stream>>>(p.f2W, f2p, 64);
    make_wkr<<<(Kd + 1 + 7) / 8, 256, 0, stream>>>(p.WK, p.bK, p.att, wkr, Kd);

    gemm_bf16_n128<<<(n_src + 127) / 128, 256, 0, stream>>>(Hb, WVp, p.bV, Vff, n_src, Kd2, 0);
    gemv_alpha<<<(n_src + 7) / 8, 256, 0, stream>>>(Hf, wkr, abuf, n_src, Kd);

    fill_f32<<<(n_tgt + 255) / 256, 256, 0, stream>>>(mx, -INFINITY, n_tgt);
    fill_f32<<<(n_tgt + 255) / 256, 256, 0, stream>>>(den, 0.0f, n_tgt);
    fill_f32<<<(n_tgt * 128 + 255) / 256, 256, 0, stream>>>(agg, 0.0f, n_tgt * 128);

    seg_max<<<(NNZ + 255) / 256, 256, 0, stream>>>(s_idx, d_idx, abuf, mx, NNZ);
    seg_den<<<(NNZ + 255) / 256, 256, 0, stream>>>(s_idx, d_idx, abuf, mx, den, NNZ);
    scatter_v<<<(NNZ + 7) / 8, 256, 0, stream>>>(s_idx, d_idx, abuf, mx, den, Vff, agg, NNZ);

    epiA<<<(n_tgt + 7) / 8, 256, 0, stream>>>(agg, p.att, p.ln0g, p.ln0b, outb, n_tgt);
    gemm_bf16_n128<<<(n_tgt + 127) / 128, 256, 0, stream>>>(outb, f1p, p.f1b, t1f, n_tgt, 64, 1);
    pack_acts<<<(n_tgt * 32 + 255) / 256, 256, 0, stream>>>(t1f, t1b, n_tgt * 32);
    gemm_bf16_n128<<<(n_tgt + 127) / 128, 256, 0, stream>>>(t1b, f2p, p.f2b, Vff, n_tgt, 64, 1);
    epiB<<<(n_tgt + 7) / 8, 256, 0, stream>>>(agg, Vff, p.ln1g, p.ln1b, out_f, out_b, n_tgt);
  };

  // hop 0: v2e0  (nodes -> hedges), input x fp32/bf16, Kd=512
  hop(P[0], e_src, dsts, N_NODES, N_HEDGES, x, xb, 512, hE, hEb);
  // hop 1: e2v0  (hedges -> nodes), reversed indices, Kd=128
  hop(P[1], dsts, e_src, N_HEDGES, N_NODES, hE, hEb, 128, hA, hAb);
  // hop 2: v2e1
  hop(P[2], e_src, dsts, N_NODES, N_HEDGES, hA, hAb, 128, hE, hEb);
  // hop 3: e2v1 -> final output (fp32, [N_NODES,128])
  hop(P[3], dsts, e_src, N_HEDGES, N_NODES, hE, hEb, 128, (float*)d_out, nullptr);
}
